// LeNet_5_56599079027371
// MI455X (gfx1250) — matrix operand order repaired and checked by reading, not compiled, not run on hardware
//
#include <hip/hip_runtime.h>
#include <math.h>

// ---------------------------------------------------------------------------
// LeNet-5 forward, B=16384, on CDNA5 WMMA (v_wmma_f32_16x16x32_f16, wave32).
// im2col GEMMs with ISA-layout fragments (2x global_load_b128 per fragment),
// image-aligned M tiling (no per-row divmods), vectorized b128 stores,
// branch-free v_exp_f32/v_rcp_f32 tanh.
// ---------------------------------------------------------------------------

typedef __attribute__((ext_vector_type(16))) _Float16 v16h;
typedef __attribute__((ext_vector_type(8)))  _Float16 v8h;
typedef __attribute__((ext_vector_type(4)))  _Float16 v4h;
typedef __attribute__((ext_vector_type(8)))  float    v8f;

#define BATCH 16384

__constant__ unsigned c3_mask_bits[16] = {
    7u, 14u, 28u, 56u, 49u, 35u, 15u, 30u,
    60u, 57u, 51u, 39u, 27u, 54u, 45u, 63u
};

// tanh_act(x) = 1.7159 * tanh(2/3 x), branch-free.
__device__ __forceinline__ float tanh_act(float x) {
    float y  = 0.66666667f * x;
    float ay = __builtin_fabsf(y);
    float e  = __expf(-2.0f * ay);
    float t  = (1.0f - e) * __builtin_amdgcn_rcpf(1.0f + e);
    return 1.7159f * __builtin_copysignf(t, y);
}

// A-matrix element->K mapping: e=0..7 -> K = kh*8+e ; e=8..15 -> K = 16+kh*8+(e-8)
__device__ __forceinline__ int a_koff(int e, int kh) {
    return ((e & 8) ? 16 : 0) + kh * 8 + (e & 7);
}

__device__ __forceinline__ v8f wmma_f16(v16h a, v16h b, v8f c) {
    return __builtin_amdgcn_wmma_f32_16x16x32_f16(
        false, a, false, b, (short)0, c, false, false);
}

// B fragment from transposed weights (N x Kpad row-major): K = kh*8 + e ; 16 + kh*8 + (e-8).
__device__ __forceinline__ v16h load_bfrag(const _Float16* __restrict__ Bt,
                                           int Kpad, int n, int kbase, int kh) {
    const _Float16* br = Bt + (size_t)n * Kpad + kbase + kh * 8;
    v8h lo = *(const v8h*)br;
    v8h hi = *(const v8h*)(br + 16);
    v16h b;
#pragma unroll
    for (int e = 0; e < 8; ++e) { b[e] = lo[e]; b[e + 8] = hi[e]; }
    return b;
}

// A fragment from K-padded row-major activations.
__device__ __forceinline__ v16h load_afrag(const _Float16* __restrict__ A,
                                           int Kpad, int row, int kbase, int kh) {
    const _Float16* ar = A + (size_t)row * Kpad + kbase + kh * 8;
    v8h lo = *(const v8h*)ar;
    v8h hi = *(const v8h*)(ar + 16);
    v16h a;
#pragma unroll
    for (int e = 0; e < 8; ++e) { a[e] = lo[e]; a[e + 8] = hi[e]; }
    return a;
}

// ---------------------------------------------------------------------------
// Weight packing: f32 -> f16, transposed (N x Kpad), zero-padded.
// ---------------------------------------------------------------------------
__global__ void pack_weights_kernel(const float* __restrict__ w1,
                                    const float* __restrict__ w3,
                                    const float* __restrict__ w5,
                                    const float* __restrict__ f6w,
                                    _Float16* __restrict__ B1t,   // 16 x 32
                                    _Float16* __restrict__ B3t,   // 16 x 160
                                    _Float16* __restrict__ B5t,   // 128 x 416
                                    _Float16* __restrict__ B6t)   // 96 x 128
{
    const int tid = threadIdx.x;
    for (int i = tid; i < 16 * 32; i += 256) {
        int n = i >> 5, k = i & 31;
        B1t[i] = (_Float16)((k < 25 && n < 6) ? w1[n * 25 + k] : 0.f);
    }
    for (int i = tid; i < 16 * 160; i += 256) {
        int n = i / 160, k = i % 160;
        float v = 0.f;
        if (k < 150) {
            int c = k / 25, t = k % 25;
            if ((c3_mask_bits[n] >> c) & 1u) v = w3[(n * 6 + c) * 25 + t];
        }
        B3t[i] = (_Float16)v;
    }
    for (int i = tid; i < 128 * 416; i += 256) {
        int n = i / 416, k = i % 416;
        B5t[i] = (_Float16)((k < 400 && n < 120) ? w5[n * 400 + k] : 0.f);
    }
    for (int i = tid; i < 96 * 128; i += 256) {
        int n = i >> 7, k = i & 127;
        B6t[i] = (_Float16)((k < 120 && n < 84) ? f6w[n * 120 + k] : 0.f);
    }
}

// ---------------------------------------------------------------------------
// C1: conv 1->6 5x5, 32x32 -> 28x28, bias + tanh.
// M = B*784 = B*49 tiles of 16 (tiles never cross an image).
// ---------------------------------------------------------------------------
__global__ void conv1_kernel(const float* __restrict__ x,
                             const _Float16* __restrict__ B1t,
                             const float* __restrict__ b1,
                             _Float16* __restrict__ out)   // (B,6,784) f16
{
    const int lane = threadIdx.x & 31;
    const int wave = threadIdx.x >> 5;
    const int m    = lane & 15;
    const int kh   = lane >> 4;
    const int n    = lane & 15;

    const int mtile = __builtin_amdgcn_readfirstlane((int)(blockIdx.x * 8u + wave));
    const int img   = mtile / 49;          // wave-uniform (SALU)
    const int p0    = (mtile % 49) * 16;   // first output pixel of this tile

    // A fragment: im2col gather (f32 -> f16); row p = p0 + m always < 784.
    const unsigned p  = (unsigned)(p0 + m);
    const unsigned oy = p / 28u, ox = p % 28u;
    const float* xi = x + (size_t)img * 1024u + (size_t)oy * 32u + ox;
    v16h a;
#pragma unroll
    for (int e = 0; e < 16; ++e) {
        int k = a_koff(e, kh);
        _Float16 v = (_Float16)0.f;
        if (k < 25) { int dy = k / 5, dx = k % 5; v = (_Float16)xi[dy * 32 + dx]; }
        a[e] = v;
    }
    v16h bf = load_bfrag(B1t, 32, n, 0, kh);

    v8f acc = {};
    acc = wmma_f16(a, bf, acc);

    // Lane's 8 outputs are rows p0+kh*8+0..7 of channel (img,n): one b128 store.
    if (n < 6) {
        const float bias = b1[n];
        v8h o;
#pragma unroll
        for (int r = 0; r < 8; ++r) o[r] = (_Float16)tanh_act(acc[r] + bias);
        _Float16* op = out + ((size_t)img * 6u + (unsigned)n) * 784u + p0 + kh * 8;
        *(v8h*)op = o;
    }
}

// ---------------------------------------------------------------------------
// Pooling: mean of 4 consecutive flat elements + affine + tanh.
// in channel stride ICS halfs; out[b*RS + c*T + t].
// ---------------------------------------------------------------------------
template <int C, int T, int RS, int ICS>
__global__ void pool_kernel(const _Float16* __restrict__ in,
                            _Float16* __restrict__ out,
                            const float* __restrict__ sw,
                            const float* __restrict__ sb,
                            unsigned total)
{
    unsigned idx = blockIdx.x * 256u + threadIdx.x;
    if (idx >= total) return;
    unsigned t  = idx % (unsigned)T;
    unsigned bc = idx / (unsigned)T;
    unsigned c  = bc % (unsigned)C;
    unsigned b  = bc / (unsigned)C;
    v4h q = *(const v4h*)(in + (size_t)bc * ICS + (size_t)t * 4u);
    float mavg = 0.25f * ((float)q[0] + (float)q[1] + (float)q[2] + (float)q[3]);
    out[(size_t)b * RS + (size_t)c * T + t] =
        (_Float16)tanh_act(sw[c] * mavg + sb[c]);
}

// Zero the K pad (400..415) of the S4 activation rows.
__global__ void s4_pad_kernel(_Float16* __restrict__ s4, unsigned total) {
    unsigned idx = blockIdx.x * 256u + threadIdx.x;   // total = B*16
    if (idx >= total) return;
    unsigned row = idx >> 4, k = idx & 15;
    s4[(size_t)row * 416u + 400u + k] = (_Float16)0.f;
}

// ---------------------------------------------------------------------------
// C3: masked conv 6->16 5x5, 14x14 -> 10x10 (mask baked into B3t).
// M padded to 112 rows/image (7 tiles of 16); channel image stride 104 halfs.
// K = 150 (pad 160, 5 k-tiles), N = 16.
// ---------------------------------------------------------------------------
__global__ void conv3_kernel(const _Float16* __restrict__ s2,  // (B,6,196)
                             const _Float16* __restrict__ B3t, // 16 x 160
                             const float* __restrict__ b3,
                             _Float16* __restrict__ out)       // (B,16,104)
{
    const int lane = threadIdx.x & 31;
    const int wave = threadIdx.x >> 5;
    const int m    = lane & 15;
    const int kh   = lane >> 4;
    const int n    = lane & 15;

    const int mtile = __builtin_amdgcn_readfirstlane((int)(blockIdx.x * 8u + wave));
    const int img   = mtile / 7;          // wave-uniform
    const int p0    = (mtile % 7) * 16;   // 0,16,...,96

    const unsigned p  = (unsigned)(p0 + m);
    const unsigned pc = (p < 100u) ? p : 0u;   // clamp padded rows (discarded)
    const unsigned oy = pc / 10u, ox = pc % 10u;
    const _Float16* si = s2 + (size_t)img * (6u * 196u) + (size_t)oy * 14u + ox;

    v8f acc = {};
#pragma unroll
    for (int kt = 0; kt < 5; ++kt) {
        v16h a;
#pragma unroll
        for (int e = 0; e < 16; ++e) {
            int k = kt * 32 + a_koff(e, kh);
            _Float16 v = (_Float16)0.f;
            if (kt < 4 || k < 150) {   // k < 150 guaranteed for kt<4
                int c = k / 25, t = k % 25, dy = t / 5, dx = t % 5;
                v = si[c * 196 + dy * 14 + dx];
            }
            a[e] = v;
        }
        v16h bf = load_bfrag(B3t, 160, n, kt * 32, kh);
        acc = wmma_f16(a, bf, acc);
    }

    const float bias = b3[n];
    v8h o;
#pragma unroll
    for (int r = 0; r < 8; ++r) o[r] = (_Float16)tanh_act(acc[r] + bias);

    _Float16* op = out + ((size_t)img * 16u + (unsigned)n) * 104u + p0 + kh * 8;
    if (p0 < 96) {
        *(v8h*)op = o;                 // rows p0+kh*8+0..7 all < 100
    } else {
#pragma unroll
        for (int r = 0; r < 8; ++r)
            if (p0 + kh * 8 + r < 100) op[r] = o[r];
    }
}

// ---------------------------------------------------------------------------
// GEMM + bias + tanh for C5 / F6 (compile-time shapes).
// A: (M, KPAD) f16 pads zeroed. Bt: (Npad, KPAD) f16 pads zeroed.
// One wave per 16x16 output tile.
// ---------------------------------------------------------------------------
template <int KPAD, int KTILES, int NTN, int NREAL, int LDO>
__global__ void gemm_tanh_kernel(const _Float16* __restrict__ A,
                                 const _Float16* __restrict__ Bt,
                                 const float* __restrict__ bias,
                                 _Float16* __restrict__ out)
{
    const int lane = threadIdx.x & 31;
    const int wave = threadIdx.x >> 5;
    const int m    = lane & 15;
    const int kh   = lane >> 4;
    const int n    = lane & 15;

    const int id    = __builtin_amdgcn_readfirstlane((int)(blockIdx.x * 8u + wave));
    const int mtile = id / NTN;
    const int ntile = id % NTN;
    const int row0  = mtile * 16;

    v8f acc = {};
#pragma unroll
    for (int kt = 0; kt < KTILES; ++kt) {
        v16h a  = load_afrag(A, KPAD, row0 + m, kt * 32, kh);
        v16h bf = load_bfrag(Bt, KPAD, ntile * 16 + n, kt * 32, kh);
        acc = wmma_f16(a, bf, acc);
    }

    const int col = ntile * 16 + n;
    const float bv = (col < NREAL) ? bias[col] : 0.f;
#pragma unroll
    for (int r = 0; r < 8; ++r) {
        int rr = row0 + r + kh * 8;
        float v = (col < NREAL) ? tanh_act(acc[r] + bv) : 0.f;
        out[(size_t)rr * LDO + col] = (_Float16)v;
    }
}

// ---------------------------------------------------------------------------
// RBF output: squared euclidean distance to each of 10 class bitmaps.
// ---------------------------------------------------------------------------
__global__ void rbf_kernel(const _Float16* __restrict__ h,   // (B,96), pad 0
                           const float* __restrict__ rbf,    // (10,84)
                           float* __restrict__ out)          // (B,10)
{
    unsigned idx = blockIdx.x * 256u + threadIdx.x;   // < B*10 exactly
    unsigned b = idx / 10u, k = idx % 10u;
    const _Float16* hp = h + (size_t)b * 96u;
    const float*    wp = rbf + (size_t)k * 84u;
    float s = 0.f;
#pragma unroll
    for (int j0 = 0; j0 < 84; j0 += 4) {
        v4h q = *(const v4h*)(hp + j0);
#pragma unroll
        for (int j = 0; j < 4; ++j) {
            float d = (float)q[j] - wp[j0 + j];
            s += d * d;
        }
    }
    out[idx] = s;
}

// ---------------------------------------------------------------------------
// Launch
// ---------------------------------------------------------------------------
extern "C" void kernel_launch(void* const* d_in, const int* in_sizes, int n_in,
                              void* d_out, int out_size, void* d_ws, size_t ws_size,
                              hipStream_t stream) {
    (void)in_sizes; (void)n_in; (void)out_size; (void)ws_size;
    const float* x     = (const float*)d_in[0];
    const float* w1    = (const float*)d_in[1];
    const float* b1    = (const float*)d_in[2];
    const float* s2_w  = (const float*)d_in[3];
    const float* s2_b  = (const float*)d_in[4];
    const float* w3    = (const float*)d_in[5];
    const float* b3    = (const float*)d_in[6];
    const float* s4_w  = (const float*)d_in[7];
    const float* s4_b  = (const float*)d_in[8];
    const float* w5    = (const float*)d_in[9];
    const float* b5    = (const float*)d_in[10];
    const float* f6_w  = (const float*)d_in[11];
    const float* f6_b  = (const float*)d_in[12];
    const float* rbf_w = (const float*)d_in[13];
    float* out = (float*)d_out;

    // Workspace layout (f16 elements; all offsets multiples of 8 -> 16B aligned)
    const size_t B = BATCH;
    size_t off = 0;
    _Float16* ws  = (_Float16*)d_ws;
    _Float16* c1w = ws + off; off += B * 6 * 784;   // (B,6,784)
    _Float16* s2w = ws + off; off += B * 6 * 196;   // (B,6,196)
    _Float16* c3w = ws + off; off += B * 16 * 104;  // (B,16,104) padded stride
    _Float16* s4w = ws + off; off += B * 416;       // (B,416) K-padded
    _Float16* c5w = ws + off; off += B * 128;       // (B,128) pad cols zeroed
    _Float16* f6o = ws + off; off += B * 96;        // (B,96)  pad cols zeroed
    _Float16* B1t = ws + off; off += 16 * 32;
    _Float16* B3t = ws + off; off += 16 * 160;
    _Float16* B5t = ws + off; off += 128 * 416;
    _Float16* B6t = ws + off; off += 96 * 128;

    // 0) pack weights (transposed, f16, padded)
    pack_weights_kernel<<<1, 256, 0, stream>>>(w1, w3, w5, f6_w, B1t, B3t, B5t, B6t);

    // 1) C1: B*49 tiles / 8 waves -> 100352 blocks
    conv1_kernel<<<(B * 49) / 8, 256, 0, stream>>>(x, B1t, b1, c1w);

    // 2) S2 pool: B*6*196 outputs
    {
        unsigned total = (unsigned)(B * 6 * 196);
        pool_kernel<6, 196, 6 * 196, 784>
            <<<total / 256, 256, 0, stream>>>(c1w, s2w, s2_w, s2_b, total);
    }

    // 3) C3: B*7 tiles / 8 waves -> 14336 blocks
    conv3_kernel<<<(B * 7) / 8, 256, 0, stream>>>(s2w, B3t, b3, c3w);

    // 4) S4 pool into K-padded (B,416) rows + zero the pad tail
    {
        unsigned total = (unsigned)(B * 16 * 25);
        pool_kernel<16, 25, 416, 104>
            <<<total / 256, 256, 0, stream>>>(c3w, s4w, s4_w, s4_b, total);
        unsigned ptotal = (unsigned)(B * 16);
        s4_pad_kernel<<<ptotal / 256, 256, 0, stream>>>(s4w, ptotal);
    }

    // 5) C5: (B x 416) * (416 x 120pad128) -> (B,128); 8 N-tiles
    gemm_tanh_kernel<416, 13, 8, 120, 128>
        <<<(B / 16) * 8 / 8, 256, 0, stream>>>(s4w, B5t, b5, c5w);

    // 6) F6: (B x 128) * (128 x 84pad96) -> (B,96); 6 N-tiles
    gemm_tanh_kernel<128, 4, 6, 84, 96>
        <<<(B / 16) * 6 / 8, 256, 0, stream>>>(c5w, B6t, f6_b, f6o);

    // 7) RBF distances -> (B,10) f32
    rbf_kernel<<<(B * 10) / 256, 256, 0, stream>>>(f6o, rbf_w, out);
}
